// GraphConvNetAID_68959994904604
// MI455X (gfx1250) — matrix operand order repaired and checked by reading, not compiled, not run on hardware
//
#include <hip/hip_runtime.h>

// ---------------------------------------------------------------------------
// Types for WMMA fragments (wave32, gfx1250)
// ---------------------------------------------------------------------------
typedef __attribute__((ext_vector_type(16))) _Float16 v16h;
typedef __attribute__((ext_vector_type(8)))  float    v8f;

union AH { v16h v; _Float16 h[16]; };
struct __attribute__((aligned(16))) H8 { _Float16 x[8]; };

__device__ __forceinline__ v8f wmma_acc(v16h a, v16h b, v8f c) {
  // D = A(16x32 f16) * B(32x16 f16) + C(16x16 f32)
  return __builtin_amdgcn_wmma_f32_16x16x32_f16(
      /*neg_a=*/false, a, /*neg_b=*/false, b,
      /*c_mod=*/(short)0, c, /*reuse_a=*/false, /*reuse_b=*/false);
}

__device__ __forceinline__ void load_or_zero(AH& a, const _Float16* p, int kb, bool ok) {
  if (ok) {
    *(H8*)&a.h[0] = *(const H8*)(p + kb);
    *(H8*)&a.h[8] = *(const H8*)(p + kb + 16);
  } else {
#pragma unroll
    for (int t = 0; t < 16; ++t) a.h[t] = (_Float16)0.f;
  }
}

// ---------------------------------------------------------------------------
// Weight prep: W (2, Co, Ci, 2) f32  ->  three B-fragments in f16, tile layout
//   dst[((mat*KT + kt)*NT + nt)*512 + lane*16 + h]
//   B layout: lane 0-15 -> K = kt*32 + (h&7) + 16*(h>>3), N = lane&15
//             lane 16-31 -> K = kt*32 + 8 + (h&7) + 16*(h>>3), N = lane&15
//   mat 0: Ws = W[0,:,:,0] + W[1,:,:,0]; mat 1: Wh = W[0,:,:,1]; mat 2: Wv = W[1,:,:,1]
// ---------------------------------------------------------------------------
__global__ void prep_weights(const float* __restrict__ W, _Float16* __restrict__ dst,
                             int Ci, int Co, int KT, int NT) {
  long total = 3L * KT * NT * 512;
  long stride = (long)gridDim.x * blockDim.x;
  for (long idx = (long)blockIdx.x * blockDim.x + threadIdx.x; idx < total; idx += stride) {
    int h    = (int)(idx & 15);
    long t   = idx >> 4;
    int lane = (int)(t & 31); t >>= 5;
    int nt   = (int)(t % NT); t /= NT;
    int kt   = (int)(t % KT); t /= KT;
    int mat  = (int)t;
    int n = nt * 16 + (lane & 15);
    int k = kt * 32 + ((lane >> 4) << 3) + (h & 7) + ((h >> 3) << 4);
    float v = 0.f;
    if (n < Co && k < Ci) {
      long base = ((long)n * Ci + k) * 2;       // graph 0
      long g1   = (long)Co * Ci * 2;            // graph 1 offset
      if      (mat == 0) v = W[base] + W[g1 + base];
      else if (mat == 1) v = W[base + 1];
      else               v = W[g1 + base + 1];
    }
    dst[idx] = (_Float16)v;
  }
}

// ---------------------------------------------------------------------------
// Layer 1: input NCHW f32 (B,3,H,W), K padded 3->32, output NHWC f16 pre-pool
// One wave per 16-pixel x 16-channel tile.
// ---------------------------------------------------------------------------
__global__ __launch_bounds__(32)
void gconv_l1(const float* __restrict__ x, const _Float16* __restrict__ wp,
              const float* __restrict__ bias, _Float16* __restrict__ out,
              int H, int W, int Co, int NT) {
  const int lane = threadIdx.x;
  const int jt = blockIdx.x, i = blockIdx.y;
  const int b = blockIdx.z / NT, nt = blockIdx.z % NT;
  const int g = lane >> 4;
  const int j = jt * 16 + (lane & 15);
  const long HW = (long)H * W;

  AH aS, aH, aV;
#pragma unroll
  for (int h = 0; h < 16; ++h) {
    int K = g * 8 + (h & 7) + ((h >> 3) << 4);
    float vs = 0.f, vh = 0.f, vv = 0.f;
    if (K < 3) {
      const float* plane = x + ((long)b * 3 + K) * HW;
      if (j < W)                vs = plane[(long)i * W + j];
      if (j + 1 < W)            vh = plane[(long)i * W + j + 1];
      if (j < W && i + 1 < H)   vv = plane[(long)(i + 1) * W + j];
    }
    aS.h[h] = (_Float16)vs; aH.h[h] = (_Float16)vh; aV.h[h] = (_Float16)vv;
  }

  const long matStride = (long)NT * 512;   // KT == 1
  const _Float16* w0 = wp + (long)nt * 512 + lane * 16;
  v16h bS = *(const v16h*)(w0);
  v16h bH = *(const v16h*)(w0 + matStride);
  v16h bV = *(const v16h*)(w0 + 2 * matStride);

  v8f acc = {};
  acc = wmma_acc(aS.v, bS, acc);
  acc = wmma_acc(aH.v, bH, acc);
  acc = wmma_acc(aV.v, bV, acc);

  int n = nt * 16 + (lane & 15);
  float bv = (n < Co) ? bias[n] : 0.f;
#pragma unroll
  for (int r = 0; r < 8; ++r) {
    int jj = jt * 16 + r + g * 8;
    if (jj < W && n < Co) {
      float v = acc[r] + bv;
      v = v > 0.f ? v : 0.f;
      out[(((long)b * H + i) * W + jj) * Co + n] = (_Float16)v;
    }
  }
}

// ---------------------------------------------------------------------------
// Generic layer: input NHWC f16 (B,H,W,Ci), Ci multiple of 32.
// ---------------------------------------------------------------------------
__global__ __launch_bounds__(32)
void gconv_f16(const _Float16* __restrict__ in, const _Float16* __restrict__ wp,
               const float* __restrict__ bias, _Float16* __restrict__ out,
               int H, int W, int Ci, int Co, int NT, float scale) {
  const int lane = threadIdx.x;
  const int jt = blockIdx.x, i = blockIdx.y;
  const int b = blockIdx.z / NT, nt = blockIdx.z % NT;
  const int g = lane >> 4;
  const int j = jt * 16 + (lane & 15);
  const int KT = Ci >> 5;

  const long row = ((long)b * H + i) * W;
  const bool okS = (j < W);
  const bool okH = (j + 1 < W);
  const bool okV = (j < W) && (i + 1 < H);
  const _Float16* pS = in + (row + j) * (long)Ci;
  const _Float16* pH = in + (row + j + 1) * (long)Ci;
  const _Float16* pV = in + (row + W + j) * (long)Ci;

  const long matStride = (long)KT * NT * 512;
  v8f acc = {};
  for (int kt = 0; kt < KT; ++kt) {
    const int kb = kt * 32 + g * 8;  // halves 0-7 at kb, halves 8-15 at kb+16
    AH aS, aH, aV;
    load_or_zero(aS, pS, kb, okS);
    load_or_zero(aH, pH, kb, okH);
    load_or_zero(aV, pV, kb, okV);

    const _Float16* w0 = wp + ((long)kt * NT + nt) * 512 + lane * 16;
    v16h bS = *(const v16h*)(w0);
    v16h bH = *(const v16h*)(w0 + matStride);
    v16h bV = *(const v16h*)(w0 + 2 * matStride);
    acc = wmma_acc(aS.v, bS, acc);
    acc = wmma_acc(aH.v, bH, acc);
    acc = wmma_acc(aV.v, bV, acc);
  }

  int n = nt * 16 + (lane & 15);
  float bv = (n < Co) ? bias[n] : 0.f;
#pragma unroll
  for (int r = 0; r < 8; ++r) {
    int jj = jt * 16 + r + g * 8;
    if (jj < W && n < Co) {
      float v = acc[r] * scale + bv;
      v = v > 0.f ? v : 0.f;
      out[(((long)b * H + i) * W + jj) * Co + n] = (_Float16)v;
    }
  }
}

// ---------------------------------------------------------------------------
// 3x3 stride-2 pad-1 max pool, NHWC f16
// ---------------------------------------------------------------------------
__global__ void pool3x3(const _Float16* __restrict__ in, _Float16* __restrict__ out,
                        int H, int W, int C, int Ho, int Wo, long total) {
  long stride = (long)gridDim.x * blockDim.x;
  for (long idx = (long)blockIdx.x * blockDim.x + threadIdx.x; idx < total; idx += stride) {
    int c = (int)(idx % C);
    long t = idx / C;
    int jo = (int)(t % Wo); t /= Wo;
    int io = (int)(t % Ho);
    int b  = (int)(t / Ho);
    float m = -3.4e38f;
    for (int di = -1; di <= 1; ++di) {
      int ii = io * 2 + di;
      if (ii < 0 || ii >= H) continue;
      for (int dj = -1; dj <= 1; ++dj) {
        int jj = jo * 2 + dj;
        if (jj < 0 || jj >= W) continue;
        float v = (float)in[(((long)b * H + ii) * W + jj) * C + c];
        m = v > m ? v : m;
      }
    }
    out[idx] = (_Float16)m;
  }
}

// ---------------------------------------------------------------------------
// BatchNorm (training stats) over (B,H,W) per channel, NHWC f16 in place
// ---------------------------------------------------------------------------
__global__ void bn_zero(float* stats) {
  int i = threadIdx.x;
  if (i < 256) stats[i] = 0.f;
}

__global__ void bn_stats(const _Float16* __restrict__ data, float* __restrict__ stats,
                         int C, long total) {
  __shared__ float ssum[128];
  __shared__ float ssq[128];
  for (int c = threadIdx.x; c < C; c += blockDim.x) { ssum[c] = 0.f; ssq[c] = 0.f; }
  __syncthreads();
  long stride = (long)gridDim.x * blockDim.x;
  for (long idx = (long)blockIdx.x * blockDim.x + threadIdx.x; idx < total; idx += stride) {
    int c = (int)(idx % C);
    float v = (float)data[idx];
    atomicAdd(&ssum[c], v);
    atomicAdd(&ssq[c], v * v);
  }
  __syncthreads();
  for (int c = threadIdx.x; c < C; c += blockDim.x) {
    atomicAdd(&stats[c], ssum[c]);
    atomicAdd(&stats[128 + c], ssq[c]);
  }
}

__global__ void bn_apply(_Float16* __restrict__ data, const float* __restrict__ stats,
                         const float* __restrict__ gam, const float* __restrict__ bet,
                         int C, long total, float invN) {
  long stride = (long)gridDim.x * blockDim.x;
  for (long idx = (long)blockIdx.x * blockDim.x + threadIdx.x; idx < total; idx += stride) {
    int c = (int)(idx % C);
    float mean = stats[c] * invN;
    float var  = stats[128 + c] * invN - mean * mean;
    float x = (float)data[idx];
    data[idx] = (_Float16)((x - mean) * rsqrtf(var + 1e-5f) * gam[c] + bet[c]);
  }
}

// ---------------------------------------------------------------------------
// Final global mean over 5x5 spatial: (B,5,5,C) f16 -> (B,C) f32
// ---------------------------------------------------------------------------
__global__ void final_mean(const _Float16* __restrict__ in, float* __restrict__ out, int C) {
  int idx = blockIdx.x * blockDim.x + threadIdx.x;
  if (idx >= 16 * C) return;
  int b = idx / C, c = idx % C;
  float s = 0.f;
  for (int p = 0; p < 25; ++p) s += (float)in[((long)b * 25 + p) * C + c];
  out[idx] = s * (1.f / 25.f);
}

// ---------------------------------------------------------------------------
// Host orchestration
// ---------------------------------------------------------------------------
extern "C" void kernel_launch(void* const* d_in, const int* in_sizes, int n_in,
                              void* d_out, int out_size, void* d_ws, size_t ws_size,
                              hipStream_t stream) {
  static const int CI[7] = {3, 32, 64, 64, 128, 128, 64};
  static const int CO[7] = {32, 64, 64, 128, 128, 64, 30};
  static const int HS[8] = {600, 300, 150, 75, 38, 19, 10, 5};
  const int B = 16;

  const float* x = (const float*)d_in[0];
  const float* Wt[7]; const float* bs[7];
  for (int l = 0; l < 7; ++l) { Wt[l] = (const float*)d_in[1 + 2 * l]; bs[l] = (const float*)d_in[2 + 2 * l]; }
  const float* gm[6]; const float* bt[6];
  for (int l = 0; l < 6; ++l) { gm[l] = (const float*)d_in[15 + 2 * l]; bt[l] = (const float*)d_in[16 + 2 * l]; }
  float* out = (float*)d_out;

  char* ws = (char*)d_ws;
  _Float16* wprep = (_Float16*)ws;                      // <=1 MB of packed weight tiles
  float* stats = (float*)(ws + (size_t)(1u << 20));     // 256 f32 BN accumulators
  _Float16* buf0 = (_Float16*)(ws + (size_t)(2u << 20));            // pre-pool (max 369 MB)
  size_t buf0_bytes = (size_t)B * 600 * 600 * 32 * sizeof(_Float16);
  _Float16* buf1 = (_Float16*)(ws + (size_t)(2u << 20) +
                               ((buf0_bytes + 255) & ~(size_t)255)); // post-pool (max 93 MB)

  // Pack weights for all layers
  _Float16* wp[7];
  size_t woff = 0;
  for (int l = 0; l < 7; ++l) {
    int KT = (CI[l] + 31) / 32, NT = (CO[l] + 15) / 16;
    wp[l] = wprep + woff;
    long total = 3L * KT * NT * 512;
    woff += (size_t)total;
    int blocks = (int)((total + 255) / 256);
    prep_weights<<<blocks, 256, 0, stream>>>(Wt[l], wp[l], CI[l], CO[l], KT, NT);
  }

  for (int l = 0; l < 7; ++l) {
    const int H = HS[l], W = H;
    const int Ho = HS[l + 1], Wo = Ho;
    const int Ci = CI[l], Co = CO[l];
    const int NT = (Co + 15) / 16;

    dim3 grid((W + 15) / 16, H, B * NT);
    if (l == 0)
      gconv_l1<<<grid, 32, 0, stream>>>(x, wp[0], bs[0], buf0, H, W, Co, NT);
    else
      gconv_f16<<<grid, 32, 0, stream>>>(buf1, wp[l], bs[l], buf0, H, W, Ci, Co, NT,
                                         (l == 6) ? 0.5f : 1.0f);

    long totalO = (long)B * Ho * Wo * Co;
    int pb = (int)((totalO + 255) / 256); if (pb > (1 << 18)) pb = 1 << 18;
    pool3x3<<<pb, 256, 0, stream>>>(buf0, buf1, H, W, Co, Ho, Wo, totalO);

    if (l < 6) {
      bn_zero<<<1, 256, 0, stream>>>(stats);
      int sb = (int)((totalO + 255) / 256); if (sb > 4096) sb = 4096;
      bn_stats<<<sb, 256, 0, stream>>>(buf1, stats, Co, totalO);
      float invN = 1.0f / (float)(B * Ho * Wo);
      int ab = (int)((totalO + 255) / 256); if (ab > (1 << 18)) ab = 1 << 18;
      bn_apply<<<ab, 256, 0, stream>>>(buf1, stats, gm[l], bt[l], Co, totalO, invN);
    } else {
      final_mean<<<(B * Co + 255) / 256, 256, 0, stream>>>(buf1, out, Co);
    }
  }
}
